// AUGRU_51024211476555
// MI455X (gfx1250) — matrix operand order repaired and checked by reading, not compiled, not run on hardware
//
#include <hip/hip_runtime.h>
#include <hip/hip_bf16.h>
#include <math.h>

// AUGRU: B=256, S=128, D=512.
// softmax over a length-1 axis == 1.0 exactly -> attention branch (x@Wa, item)
// is dead code and dropped with no numeric change.
//
//   K1 pack_weights : fp32 [K,N] -> bf16 [N,K] for Wu,Wr,Wh,Uu,Ur,Uh (ws)
//   K2 xw_gemm      : Gx[g] = x @ W_g (+bias), bf16 in ws      -- bf16 WMMA
//   K3 recurrent    : 16-row batch tile per WG, S-loop in-kernel, h@U* via
//                     bf16 WMMA, gates fused in registers, h state in LDS.
// Fragment loads are software-pipelined (B 4-deep rotation, A double buffer).

#define B_  256
#define S_  128
#define D_  512
#define M_  (B_ * S_)          // 32768 flattened x rows

typedef __attribute__((ext_vector_type(16))) __bf16 v16bf;
typedef __attribute__((ext_vector_type(8)))  __bf16 v8bf;
typedef __attribute__((ext_vector_type(8)))  float  v8f;
typedef __attribute__((ext_vector_type(4)))  float  v4f;

__device__ __forceinline__ float sigmoid_(float x) {
    return 1.0f / (1.0f + __expf(-x));
}
__device__ __forceinline__ float tanh_(float x) {
    // 2*sigmoid(2x) - 1 : one v_exp_f32, no libcall
    return 2.0f / (1.0f + __expf(-2.0f * x)) - 1.0f;
}

// ---------------------------------------------------------------------------
// A-fragment (16x32 bf16, MxK) built from an fp32 source (global or LDS).
// ISA layout: lanes 0-15 hold M=lane, K = {0..7, 16..23};
//             lanes 16-31 hold M=lane-16, K = {8..15, 24..31}.
// ---------------------------------------------------------------------------
__device__ __forceinline__ v16bf load_a_frag_f32(const float* base, int ld,
                                                 int k0, int lane) {
    const int m    = lane & 15;
    const int half = lane >> 4;          // 0 or 1
    const float* p = base + m * ld + k0 + half * 8;
    const float* q = base + m * ld + k0 + 16 + half * 8;
    v4f f0 = *(const v4f*)(p);
    v4f f1 = *(const v4f*)(p + 4);
    v4f f2 = *(const v4f*)(q);
    v4f f3 = *(const v4f*)(q + 4);
    v16bf a;
#pragma unroll
    for (int i = 0; i < 4; ++i) {
        a[i]      = (__bf16)f0[i];
        a[4 + i]  = (__bf16)f1[i];
        a[8 + i]  = (__bf16)f2[i];
        a[12 + i] = (__bf16)f3[i];
    }
    return a;
}

// ---------------------------------------------------------------------------
// B-fragment (32x16 bf16, KxN) for flattened tile t = g*4 + j from
// pre-transposed bf16 weights wbase[g][N][K].
// ISA layout: lane = N (mod 16); lanes 0-15 hold K={0..7,16..23}, lanes 16-31 K={8..15,24..31}.
// ---------------------------------------------------------------------------
__device__ __forceinline__ v16bf load_b_frag_t(const __bf16* wbase, int t,
                                               int n0, int k0, int lane) {
    const int g     = t >> 2;
    const int j     = t & 3;
    const int n     = n0 + j * 16 + (lane & 15);
    const int khalf = (lane >> 4) << 3;  // 0 or 8
    const __bf16* p = wbase + (size_t)g * D_ * D_ + (size_t)n * D_ + k0 + khalf;
    v8bf lo = *(const v8bf*)(p);
    v8bf hi = *(const v8bf*)(p + 16);
    v16bf b;
#pragma unroll
    for (int i = 0; i < 8; ++i) { b[i] = lo[i]; b[8 + i] = hi[i]; }
    return b;
}

// ---------------------------------------------------------------------------
// 16x(3x64) GEMM core over K=D: 12 accumulators (t = g*4+j).
// B fragments rotate through a 4-deep buffer (3 pairs in flight under the
// WMMA chain); A is double-buffered across k-steps.
// ---------------------------------------------------------------------------
__device__ __forceinline__ void gemm12(const float* abase, int lda,
                                       const __bf16* wbase, int n0, int lane,
                                       v8f acc[12]) {
    v16bf a = load_a_frag_f32(abase, lda, 0, lane);
    for (int k0 = 0; k0 < D_; k0 += 32) {
        v16bf b0 = load_b_frag_t(wbase, 0, n0, k0, lane);
        v16bf b1 = load_b_frag_t(wbase, 1, n0, k0, lane);
        v16bf b2 = load_b_frag_t(wbase, 2, n0, k0, lane);
        v16bf b3 = load_b_frag_t(wbase, 3, n0, k0, lane);
        const int kn = (k0 + 32 < D_) ? (k0 + 32) : 0;   // wrap load, value dead
        v16bf an = load_a_frag_f32(abase, lda, kn, lane);
#pragma unroll
        for (int t = 0; t < 12; ++t) {
            v16bf bc;
            switch (t & 3) {
                case 0: bc = b0; break;
                case 1: bc = b1; break;
                case 2: bc = b2; break;
                default: bc = b3; break;
            }
            if (t + 4 < 12) {
                v16bf bn = load_b_frag_t(wbase, t + 4, n0, k0, lane);
                switch (t & 3) {
                    case 0: b0 = bn; break;
                    case 1: b1 = bn; break;
                    case 2: b2 = bn; break;
                    default: b3 = bn; break;
                }
            }
            acc[t] = __builtin_amdgcn_wmma_f32_16x16x32_bf16(
                false, a, false, bc, (short)0, acc[t], false, false);
        }
        a = an;
    }
}

// ---------------------------------------------------------------------------
// K1: transpose + fp32->bf16 for six DxD matrices. dst[mat][n*D + k] = src[k*D+n]
// ---------------------------------------------------------------------------
__global__ __launch_bounds__(256)
void pack_weights_kernel(const float* __restrict__ w0, const float* __restrict__ w1,
                         const float* __restrict__ w2, const float* __restrict__ w3,
                         const float* __restrict__ w4, const float* __restrict__ w5,
                         __bf16* __restrict__ dst) {
    const float* srcs[6] = {w0, w1, w2, w3, w4, w5};
    const int gid = blockIdx.x * 256 + threadIdx.x;   // 6*D*D threads exactly
    const int mat = gid / (D_ * D_);
    const int rem = gid % (D_ * D_);
    const int k   = rem / D_;
    const int n   = rem % D_;
    dst[(size_t)mat * D_ * D_ + (size_t)n * D_ + k] =
        (__bf16)srcs[mat][(size_t)k * D_ + n];
}

// ---------------------------------------------------------------------------
// K2: Gx[g][m, n] = (x @ W_g)[m, n] + b_g[n], stored bf16.
// One wave: 16 rows x 64 cols x 3 gates. 2048 blocks x 8 waves.
// ---------------------------------------------------------------------------
__global__ __launch_bounds__(256)
void xw_gemm_kernel(const float* __restrict__ x,
                    const __bf16* __restrict__ wt,        // WuT,WrT,WhT contiguous
                    const float* __restrict__ bu, const float* __restrict__ br,
                    const float* __restrict__ bh,
                    __bf16* __restrict__ gx) {
    const int lane  = threadIdx.x & 31;
    const int wave  = threadIdx.x >> 5;
    const int wid   = blockIdx.x * 8 + wave;
    const int mtile = wid >> 3;           // 0..2047
    const int ngrp  = wid & 7;            // 0..7
    const int m0    = mtile * 16;
    const int n0    = ngrp * 64;

    v8f z = {};
    v8f acc[12];
#pragma unroll
    for (int t = 0; t < 12; ++t) acc[t] = z;

    gemm12(x + (size_t)m0 * D_, D_, wt, n0, lane, acc);

    const float* biases[3] = {bu, br, bh};
#pragma unroll
    for (int g = 0; g < 3; ++g) {
#pragma unroll
        for (int j = 0; j < 4; ++j) {
            const int n = n0 + j * 16 + (lane & 15);
            const float bias = biases[g][n];
#pragma unroll
            for (int i = 0; i < 8; ++i) {
                const int m = m0 + i + ((lane >> 4) << 3);   // C layout: M=i or i+8
                gx[(size_t)g * M_ * D_ + (size_t)m * D_ + n] =
                    (__bf16)(acc[g * 4 + j][i] + bias);
            }
        }
    }
}

// ---------------------------------------------------------------------------
// K3: persistent recurrence. Block = 16 batch rows, 8 waves; wave owns n-range
// [wave*64, wave*64+64) for all three gates so the gate math is register-local.
// ---------------------------------------------------------------------------
#define HSTRIDE 516   // padded fp32 row stride: spreads 64-bank LDS accesses

__global__ __launch_bounds__(256, 1)
void augru_recurrent_kernel(const __bf16* __restrict__ ut,   // UuT,UrT,UhT contiguous
                            const __bf16* __restrict__ gx,   // 3 x [M_, D_] bf16
                            const float* __restrict__ h0,
                            const float* __restrict__ bu, const float* __restrict__ br,
                            const float* __restrict__ bh,
                            float* __restrict__ out,         // [B,S,D]
                            float* __restrict__ hlast) {     // [B,D]
    __shared__ float hsm[16 * HSTRIDE];

    const int lane = threadIdx.x & 31;
    const int wave = threadIdx.x >> 5;
    const int b0   = blockIdx.x * 16;
    const int n0   = wave * 64;

    // stage h0 tile into LDS (fp32 master copy of the state)
    for (int idx = threadIdx.x; idx < 16 * D_; idx += 256) {
        const int m = idx >> 9;          // /512
        const int n = idx & 511;
        hsm[m * HSTRIDE + n] = h0[(size_t)(b0 + m) * D_ + n];
    }
    __syncthreads();

    // per-lane biases for owned columns (constant over the S loop)
    float bs[3][4];
#pragma unroll
    for (int j = 0; j < 4; ++j) {
        const int n = n0 + j * 16 + (lane & 15);
        bs[0][j] = bu[n];
        bs[1][j] = br[n];
        bs[2][j] = bh[n];
    }

    const v8f z = {};
    for (int s = 0; s < S_; ++s) {
        // Prefetch this step's full gate-preactivation tile: one aligned 128B
        // line per (gate,row) segment; lane -> row (lane&15). Issued before
        // the K-loop so the lines land during the 192 WMMAs.
        {
            const size_t rowi = ((size_t)(b0 + (lane & 15)) * S_ + s) * D_ + n0;
#pragma unroll
            for (int g = 0; g < 3; ++g)
                __builtin_prefetch(gx + (size_t)g * M_ * D_ + rowi, 0, 3);
        }

        v8f acc[12];
#pragma unroll
        for (int t = 0; t < 12; ++t) acc[t] = z;

        // h @ U* : software-pipelined bf16 WMMA over K=D
        gemm12(hsm, HSTRIDE, ut, n0, lane, acc);

        __syncthreads();   // all waves finished reading h before we overwrite it

        // fused gates, entirely in the WMMA C layout (a == 1, so u_hat == u)
#pragma unroll
        for (int j = 0; j < 4; ++j) {
            const int n = n0 + j * 16 + (lane & 15);
#pragma unroll
            for (int i = 0; i < 8; ++i) {
                const int    Mr   = i + ((lane >> 4) << 3);
                const int    brow = b0 + Mr;
                const size_t gi   = ((size_t)brow * S_ + s) * D_ + n;
                const float  gu = (float)gx[gi];
                const float  gr = (float)gx[(size_t)1 * M_ * D_ + gi];
                const float  gh = (float)gx[(size_t)2 * M_ * D_ + gi];

                const float u  = sigmoid_(gu + acc[0 * 4 + j][i] + bs[0][j]);
                const float r  = sigmoid_(gr + acc[1 * 4 + j][i] + bs[1][j]);
                const float hc = tanh_(gh + r * acc[2 * 4 + j][i] + bs[2][j]);

                const float hp = hsm[Mr * HSTRIDE + n];
                const float hn = (1.0f - u) * hp + u * hc;

                hsm[Mr * HSTRIDE + n] = hn;
                out[(size_t)brow * S_ * D_ + (size_t)s * D_ + n] = hn;
                if (s == S_ - 1) hlast[(size_t)brow * D_ + n] = hn;
            }
        }
        __syncthreads();   // h fully updated before next step's A loads
    }
}

// ---------------------------------------------------------------------------
// Host entry
// ---------------------------------------------------------------------------
extern "C" void kernel_launch(void* const* d_in, const int* in_sizes, int n_in,
                              void* d_out, int out_size, void* d_ws, size_t ws_size,
                              hipStream_t stream) {
    const float* x  = (const float*)d_in[0];
    // d_in[1] (item) and d_in[12] (Wa) are provably unused: softmax over a
    // length-1 axis is identically 1.0.
    const float* h0 = (const float*)d_in[2];
    const float* Wu = (const float*)d_in[3];
    const float* Uu = (const float*)d_in[4];
    const float* bu = (const float*)d_in[5];
    const float* Wr = (const float*)d_in[6];
    const float* Ur = (const float*)d_in[7];
    const float* br = (const float*)d_in[8];
    const float* Wh = (const float*)d_in[9];
    const float* Uh = (const float*)d_in[10];
    const float* bh = (const float*)d_in[11];

    // workspace layout (bf16): [WuT,WrT,WhT | UuT,UrT,UhT | Gx_u,Gx_r,Gx_h]
    __bf16* wt = (__bf16*)d_ws;
    __bf16* ut = wt + (size_t)3 * D_ * D_;
    __bf16* gx = wt + (size_t)6 * D_ * D_;

    float* out   = (float*)d_out;
    float* hlast = out + (size_t)B_ * S_ * D_;

    pack_weights_kernel<<<(6 * D_ * D_) / 256, 256, 0, stream>>>(
        Wu, Wr, Wh, Uu, Ur, Uh, wt);

    // 2048 m-tiles x 8 n-groups = 16384 waves / 8 waves per block
    xw_gemm_kernel<<<(M_ / 16) * 8 / 8, 256, 0, stream>>>(
        x, wt, bu, br, bh, gx);

    augru_recurrent_kernel<<<B_ / 16, 256, 0, stream>>>(
        ut, gx, h0, bu, br, bh, out, hlast);
}
